// graph_encoder_12111807775412
// MI455X (gfx1250) — matrix operand order repaired and checked by reading, not compiled, not run on hardware
//
#include <hip/hip_runtime.h>
#include <hip/hip_bf16.h>
#include <cstdint>

#define HDIM 128
#define NNODES 50000
#define NEDGES 640000
#define NGRAPH 128
#define NEMB   201
#define MT     64      // rows per block-kernel workgroup

typedef __bf16 bf16_t;
typedef __attribute__((ext_vector_type(16))) __bf16 v16b;
typedef __attribute__((ext_vector_type(8)))  float  v8f;

union AF  { v16b v; bf16_t e[16]; uint4 q[2]; };
union Q8  { uint4 q; bf16_t e[8]; };
union Q4  { uint2 u; bf16_t e[4]; };
union F4  { float4 f; float e[4]; };

// ---------------------------------------------------------------------------
// WMMA helpers (bf16 A/B, f32 accumulate). A tile is 16x32 row-major in LDS.
// Per ISA 7.12.2 (16-bit A 16x32): lanes 0-15 hold M=lane, K={0..7,16..23};
// lanes 16-31 hold M=lane-16, K={8..15,24..31}. Two ds_load_b128 per frag.
// ---------------------------------------------------------------------------
__device__ __forceinline__ v16b load_a_lds(const bf16_t* s, int ld, int lane) {
  const int m    = lane & 15;
  const int koff = (lane >> 4) << 3;   // 0 or 8
  AF a;
  a.q[0] = *(const uint4*)(s + m * ld + koff);        // K = koff .. koff+7
  a.q[1] = *(const uint4*)(s + m * ld + koff + 16);   // K = koff+16 .. koff+23
  return a.v;
}

__device__ __forceinline__ v8f wmma_bf16(v16b a, v16b b, v8f c) {
  return __builtin_amdgcn_wmma_f32_16x16x32_bf16(false, a, false, b, (short)0, c,
                                                 false, false);
}

// ---------------------------------------------------------------------------
// Convert fp32 row-major weights [K,Nc] into bf16 WMMA-B fragment-major layout
// padded to Kpad (multiple of 32). Layout: [(kt*ntn+nt)*32 + lane][16 contiguous]
// lanes 0-15 -> N=nt*16+lane, K=kt*32+{0..7,16..23}; lanes 16-31 -> N=lane-16, K=+{8..15,24..31}.
// ---------------------------------------------------------------------------
__global__ void conv_weight_frag(const float* __restrict__ w, bf16_t* __restrict__ wf,
                                 int K, int Nc, int Kpad) {
  const int idx = blockIdx.x * 256 + threadIdx.x;
  const int total = Kpad * Nc;
  if (idx >= total) return;
  const int i    = idx & 15;
  const int lane = (idx >> 4) & 31;
  const int tile = idx >> 9;
  const int ntn  = Nc >> 4;
  const int kt   = tile / ntn;
  const int nt   = tile - kt * ntn;
  const int n    = nt * 16 + (lane & 15);
  const int k    = kt * 32 + ((lane >> 4) << 3) + (i & 7) + ((i >> 3) << 4);
  wf[idx] = (bf16_t)((k < K) ? w[k * Nc + n] : 0.f);
}

__global__ void zero_f32(float* __restrict__ p, size_t n) {
  size_t i = (size_t)blockIdx.x * 256 + threadIdx.x;
  if (i < n) p[i] = 0.f;
}

// ---------------------------------------------------------------------------
// Fused 2-layer encoder MLP: out = PReLU(in@W1 + b1) @ W2 + b2
// 16 rows per workgroup, 128 threads (4 waves). Memory-bound on fp32 input.
// ---------------------------------------------------------------------------
template <int K1PAD, int KIN>
__global__ void enc_mlp_kernel(const float* __restrict__ in,
                               const bf16_t* __restrict__ w1f, const float* __restrict__ b1,
                               const float* __restrict__ aP,
                               const bf16_t* __restrict__ w2f, const float* __restrict__ b2,
                               float* __restrict__ outf, bf16_t* __restrict__ outb) {
  __shared__ __align__(16) bf16_t sA[16 * K1PAD];
  __shared__ __align__(16) bf16_t sH[16 * HDIM];
  const int tid = threadIdx.x;
  const size_t m0 = (size_t)blockIdx.x * 16;

  for (int idx = tid; idx < 16 * K1PAD; idx += 128) {
    const int r = idx / K1PAD, c = idx % K1PAD;
    float v = (c < KIN) ? in[(m0 + r) * KIN + c] : 0.f;
    sA[idx] = (bf16_t)v;
  }
  __syncthreads();

  const int wave = tid >> 5, lane = tid & 31;
  const float alpha = aP[0];
  const v16b* w1t = (const v16b*)w1f;
  v8f a0 = {}, a1 = {};
#pragma unroll
  for (int kt = 0; kt < K1PAD / 32; ++kt) {
    v16b a  = load_a_lds(sA + kt * 32, K1PAD, lane);
    v16b f0 = w1t[(kt * 8 + wave * 2 + 0) * 32 + lane];
    v16b f1 = w1t[(kt * 8 + wave * 2 + 1) * 32 + lane];
    a0 = wmma_bf16(a, f0, a0);
    a1 = wmma_bf16(a, f1, a1);
  }
  {
    const int n0 = wave * 32 + (lane & 15);
    const int n1 = n0 + 16;
    const int mb = (lane >> 4) << 3;
    const float c0 = b1[n0], c1 = b1[n1];
#pragma unroll
    for (int v = 0; v < 8; ++v) {
      float h0 = a0[v] + c0; h0 = h0 >= 0.f ? h0 : alpha * h0;
      float h1 = a1[v] + c1; h1 = h1 >= 0.f ? h1 : alpha * h1;
      sH[(v + mb) * HDIM + n0] = (bf16_t)h0;
      sH[(v + mb) * HDIM + n1] = (bf16_t)h1;
    }
  }
  __syncthreads();

  v8f o0 = {}, o1 = {};
  const v16b* w2t = (const v16b*)w2f;
#pragma unroll
  for (int kt = 0; kt < 4; ++kt) {
    v16b a  = load_a_lds(sH + kt * 32, HDIM, lane);
    v16b f0 = w2t[(kt * 8 + wave * 2 + 0) * 32 + lane];
    v16b f1 = w2t[(kt * 8 + wave * 2 + 1) * 32 + lane];
    o0 = wmma_bf16(a, f0, o0);
    o1 = wmma_bf16(a, f1, o1);
  }
  const int n0 = wave * 32 + (lane & 15);
  const int n1 = n0 + 16;
  const int mb = (lane >> 4) << 3;
  const float c0 = b2[n0], c1 = b2[n1];
#pragma unroll
  for (int v = 0; v < 8; ++v) {
    const size_t ri = (m0 + v + mb) * HDIM;
    const float r0 = o0[v] + c0, r1 = o1[v] + c1;
    if (outf) { outf[ri + n0] = r0; outf[ri + n1] = r1; }
    outb[ri + n0] = (bf16_t)r0;
    outb[ri + n1] = (bf16_t)r1;
  }
}

// ---------------------------------------------------------------------------
// Edge block: e_out = PReLU(LN([x[row]|x[col]|ea] @ W1 + b1)) @ W2 + b2
// then ea += e_out (bf16 in place) and agg[col] += e_out (f32 atomics).
// 64 edges / workgroup, 256 threads (8 waves). Each wave: GEMM1 2 N-tiles x
// 4 M-tiles (B-fragment reused 4x), GEMM2 1 N-tile x 4 M-tiles.
// ---------------------------------------------------------------------------
__global__ __launch_bounds__(256)
void edge_block_kernel(const bf16_t* __restrict__ xb,
                       bf16_t* __restrict__ eab,
                       const int* __restrict__ eidx,
                       const bf16_t* __restrict__ w1f, const float* __restrict__ b1,
                       const float* __restrict__ gam, const float* __restrict__ bet,
                       const float* __restrict__ aP,
                       const bf16_t* __restrict__ w2f, const float* __restrict__ b2,
                       float* __restrict__ agg) {
  __shared__ __align__(16) bf16_t sA[MT * 384];   // 48KB; reused as bf16 h1 [MT x 256]
  __shared__ float sPart[MT * 4], sPart2[MT * 4];
  __shared__ float sMu[MT], sRs[MT];
  __shared__ int   sRow[MT], sCol[MT];

  const int tid = threadIdx.x;
  const int e0  = blockIdx.x * MT;
  if (tid < MT) {
    sRow[tid] = eidx[e0 + tid];
    sCol[tid] = eidx[NEDGES + e0 + tid];
  }
  __syncthreads();

  // gather-concat into LDS, 16B vectors (8 bf16)
  for (int ch = tid; ch < MT * 48; ch += 256) {
    const int r  = ch / 48;
    const int cb = (ch % 48) * 8;
    const bf16_t* src;
    if (cb < 128)      src = xb  + (size_t)sRow[r] * HDIM + cb;
    else if (cb < 256) src = xb  + (size_t)sCol[r] * HDIM + (cb - 128);
    else               src = eab + (size_t)(e0 + r) * HDIM + (cb - 256);
    *(uint4*)(sA + r * 384 + cb) = *(const uint4*)src;
  }
  __syncthreads();

  const int wave = tid >> 5, lane = tid & 31;
  const v16b* w1t = (const v16b*)w1f;
  v8f acc[4][2] = {};                     // [m-tile][n-tile]
  for (int kt = 0; kt < 12; ++kt) {
    v16b f0 = w1t[(kt * 16 + wave * 2 + 0) * 32 + lane];
    v16b f1 = w1t[(kt * 16 + wave * 2 + 1) * 32 + lane];
#pragma unroll
    for (int mt = 0; mt < 4; ++mt) {
      v16b a = load_a_lds(sA + (mt * 16) * 384 + kt * 32, 384, lane);
      acc[mt][0] = wmma_bf16(a, f0, acc[mt][0]);
      acc[mt][1] = wmma_bf16(a, f1, acc[mt][1]);
    }
  }
  __syncthreads();                         // A staging no longer needed

  // h1 = acc + b1  -> LDS (bf16, pre-LN), reusing sA
  bf16_t* sH = sA;
  {
    const int n0 = wave * 32 + (lane & 15);
    const int n1 = n0 + 16;
    const int mb = (lane >> 4) << 3;
    const float c0 = b1[n0], c1 = b1[n1];
#pragma unroll
    for (int mt = 0; mt < 4; ++mt)
#pragma unroll
      for (int v = 0; v < 8; ++v) {
        const int m = mt * 16 + v + mb;
        sH[m * 256 + n0] = (bf16_t)(acc[mt][0][v] + c0);
        sH[m * 256 + n1] = (bf16_t)(acc[mt][1][v] + c1);
      }
  }
  __syncthreads();

  // LN stats: 4 threads per row, vectorized bf16 reads
  {
    const int r = tid >> 2, p = tid & 3;
    float s = 0.f, s2 = 0.f;
    const bf16_t* rp = sH + r * 256 + p * 64;
#pragma unroll
    for (int c8 = 0; c8 < 8; ++c8) {
      Q8 v; v.q = *(const uint4*)(rp + c8 * 8);
#pragma unroll
      for (int i = 0; i < 8; ++i) { float t = (float)v.e[i]; s += t; s2 += t * t; }
    }
    sPart[r * 4 + p] = s; sPart2[r * 4 + p] = s2;
  }
  __syncthreads();
  if (tid < MT) {
    const float s  = sPart [tid*4] + sPart [tid*4+1] + sPart [tid*4+2] + sPart [tid*4+3];
    const float s2 = sPart2[tid*4] + sPart2[tid*4+1] + sPart2[tid*4+2] + sPart2[tid*4+3];
    const float mu = s * (1.f / 256.f);
    sMu[tid] = mu;
    sRs[tid] = rsqrtf(s2 * (1.f / 256.f) - mu * mu + 1e-5f);
  }
  __syncthreads();

  // apply LN + PReLU in place
  const float alpha = aP[0];
  for (int idx = tid; idx < MT * 256; idx += 256) {
    const int r = idx >> 8, c = idx & 255;
    float t = ((float)sH[idx] - sMu[r]) * sRs[r] * gam[c] + bet[c];
    t = t >= 0.f ? t : alpha * t;
    sH[idx] = (bf16_t)t;
  }
  __syncthreads();

  // GEMM2: K=256, out 128 cols; wave owns N-tile = wave
  const v16b* w2t = (const v16b*)w2f;
  v8f o[4] = {};
  for (int kt = 0; kt < 8; ++kt) {
    v16b bb = w2t[(kt * 8 + wave) * 32 + lane];
#pragma unroll
    for (int mt = 0; mt < 4; ++mt) {
      v16b a = load_a_lds(sH + (mt * 16) * 256 + kt * 32, 256, lane);
      o[mt] = wmma_bf16(a, bb, o[mt]);
    }
  }

  const int n  = wave * 16 + (lane & 15);
  const int mb = (lane >> 4) << 3;
  const float c2 = b2[n];
#pragma unroll
  for (int mt = 0; mt < 4; ++mt)
#pragma unroll
    for (int v = 0; v < 8; ++v) {
      const int m = mt * 16 + v + mb;
      const size_t ei = (size_t)(e0 + m) * HDIM;
      const float r0 = o[mt][v] + c2;
      eab[ei + n] = (bf16_t)((float)eab[ei + n] + r0);
      atomicAdd(&agg[(size_t)sCol[m] * HDIM + n], r0);
    }
}

// ---------------------------------------------------------------------------
// Node block: n_out = PReLU(LN([x|agg] @ W1 + b1)) @ W2 + b2 ; x += n_out.
// 64 nodes / workgroup, 256 threads. Tail-guarded (N % 64 != 0).
// ---------------------------------------------------------------------------
__global__ __launch_bounds__(256)
void node_block_kernel(bf16_t* __restrict__ xb,
                       const float* __restrict__ agg,
                       const bf16_t* __restrict__ w1f, const float* __restrict__ b1,
                       const float* __restrict__ gam, const float* __restrict__ bet,
                       const float* __restrict__ aP,
                       const bf16_t* __restrict__ w2f, const float* __restrict__ b2,
                       float* __restrict__ xf) {
  __shared__ __align__(16) bf16_t sA[MT * 256];   // 32KB; reused as h1
  __shared__ float sPart[MT * 4], sPart2[MT * 4];
  __shared__ float sMu[MT], sRs[MT];
  const int tid = threadIdx.x;
  const size_t nd0 = (size_t)blockIdx.x * MT;

  // stage [x | agg] -> bf16 LDS
  for (int ch = tid; ch < MT * 16; ch += 256) {          // x half, 8 bf16 / chunk
    const int r = ch >> 4, c8 = (ch & 15) * 8;
    const size_t nd = nd0 + r;
    uint4 val = {0, 0, 0, 0};
    if (nd < NNODES) val = *(const uint4*)(xb + nd * HDIM + c8);
    *(uint4*)(sA + r * 256 + c8) = val;
  }
  for (int ch = tid; ch < MT * 32; ch += 256) {          // agg half, 4 f32 -> 4 bf16
    const int r = ch >> 5, c4 = (ch & 31) * 4;
    const size_t nd = nd0 + r;
    Q4 out; out.u = make_uint2(0, 0);
    if (nd < NNODES) {
      F4 v; v.f = *(const float4*)(agg + nd * HDIM + c4);
#pragma unroll
      for (int i = 0; i < 4; ++i) out.e[i] = (bf16_t)v.e[i];
    }
    *(uint2*)(sA + r * 256 + HDIM + c4) = out.u;
  }
  __syncthreads();

  const int wave = tid >> 5, lane = tid & 31;
  const v16b* w1t = (const v16b*)w1f;
  v8f acc[4][2] = {};
  for (int kt = 0; kt < 8; ++kt) {
    v16b f0 = w1t[(kt * 16 + wave * 2 + 0) * 32 + lane];
    v16b f1 = w1t[(kt * 16 + wave * 2 + 1) * 32 + lane];
#pragma unroll
    for (int mt = 0; mt < 4; ++mt) {
      v16b a = load_a_lds(sA + (mt * 16) * 256 + kt * 32, 256, lane);
      acc[mt][0] = wmma_bf16(a, f0, acc[mt][0]);
      acc[mt][1] = wmma_bf16(a, f1, acc[mt][1]);
    }
  }
  __syncthreads();

  bf16_t* sH = sA;
  {
    const int n0 = wave * 32 + (lane & 15);
    const int n1 = n0 + 16;
    const int mb = (lane >> 4) << 3;
    const float c0 = b1[n0], c1 = b1[n1];
#pragma unroll
    for (int mt = 0; mt < 4; ++mt)
#pragma unroll
      for (int v = 0; v < 8; ++v) {
        const int m = mt * 16 + v + mb;
        sH[m * 256 + n0] = (bf16_t)(acc[mt][0][v] + c0);
        sH[m * 256 + n1] = (bf16_t)(acc[mt][1][v] + c1);
      }
  }
  __syncthreads();

  {
    const int r = tid >> 2, p = tid & 3;
    float s = 0.f, s2 = 0.f;
    const bf16_t* rp = sH + r * 256 + p * 64;
#pragma unroll
    for (int c8 = 0; c8 < 8; ++c8) {
      Q8 v; v.q = *(const uint4*)(rp + c8 * 8);
#pragma unroll
      for (int i = 0; i < 8; ++i) { float t = (float)v.e[i]; s += t; s2 += t * t; }
    }
    sPart[r * 4 + p] = s; sPart2[r * 4 + p] = s2;
  }
  __syncthreads();
  if (tid < MT) {
    const float s  = sPart [tid*4] + sPart [tid*4+1] + sPart [tid*4+2] + sPart [tid*4+3];
    const float s2 = sPart2[tid*4] + sPart2[tid*4+1] + sPart2[tid*4+2] + sPart2[tid*4+3];
    const float mu = s * (1.f / 256.f);
    sMu[tid] = mu;
    sRs[tid] = rsqrtf(s2 * (1.f / 256.f) - mu * mu + 1e-5f);
  }
  __syncthreads();

  const float alpha = aP[0];
  for (int idx = tid; idx < MT * 256; idx += 256) {
    const int r = idx >> 8, c = idx & 255;
    float t = ((float)sH[idx] - sMu[r]) * sRs[r] * gam[c] + bet[c];
    t = t >= 0.f ? t : alpha * t;
    sH[idx] = (bf16_t)t;
  }
  __syncthreads();

  const v16b* w2t = (const v16b*)w2f;
  v8f o[4] = {};
  for (int kt = 0; kt < 8; ++kt) {
    v16b bb = w2t[(kt * 8 + wave) * 32 + lane];
#pragma unroll
    for (int mt = 0; mt < 4; ++mt) {
      v16b a = load_a_lds(sH + (mt * 16) * 256 + kt * 32, 256, lane);
      o[mt] = wmma_bf16(a, bb, o[mt]);
    }
  }

  const int n  = wave * 16 + (lane & 15);
  const int mb = (lane >> 4) << 3;
  const float c2 = b2[n];
#pragma unroll
  for (int mt = 0; mt < 4; ++mt)
#pragma unroll
    for (int v = 0; v < 8; ++v) {
      const size_t nd = nd0 + mt * 16 + v + mb;
      if (nd < NNODES) {
        const size_t ri = nd * HDIM;
        const float xv = xf[ri + n] + o[mt][v] + c2;
        xf[ri + n] = xv;
        xb[ri + n] = (bf16_t)xv;
      }
    }
}

// ---------------------------------------------------------------------------
// Small scalar kernels (tiny FLOP counts; memory-trivial)
// ---------------------------------------------------------------------------
__global__ void glob_enc_kernel(const float* __restrict__ glob,
                                const float* __restrict__ w1, const float* __restrict__ b1,
                                const float* __restrict__ aP,
                                const float* __restrict__ w2, const float* __restrict__ b2,
                                float* __restrict__ u) {
  const int b = blockIdx.x, j = threadIdx.x;
  __shared__ float sh[HDIM];
  const float g0 = glob[b * 2], g1 = glob[b * 2 + 1];
  float h = g0 * w1[j] + g1 * w1[HDIM + j] + b1[j];
  const float alpha = aP[0];
  h = h >= 0.f ? h : alpha * h;
  sh[j] = h;
  __syncthreads();
  float acc = b2[j];
  for (int k = 0; k < HDIM; ++k) acc += sh[k] * w2[k * HDIM + j];
  u[b * HDIM + j] = acc;
}

__global__ void pool_kernel(const float* __restrict__ xf, const int* __restrict__ batch,
                            float* __restrict__ pooled) {
  size_t i = (size_t)blockIdx.x * 256 + threadIdx.x;
  if (i >= (size_t)NNODES * HDIM) return;
  const int node = (int)(i >> 7), c = (int)(i & 127);
  atomicAdd(&pooled[(size_t)batch[node] * HDIM + c], xf[i]);
}

__global__ void graph_kernel(const float* __restrict__ u, const float* __restrict__ pooled,
                             const float* __restrict__ decw, const float* __restrict__ decb,
                             float* __restrict__ gv) {
  const int b = blockIdx.x, j = threadIdx.x;
  __shared__ float cat[2 * HDIM];
  cat[j]        = u[b * HDIM + j];
  cat[HDIM + j] = pooled[b * HDIM + j];
  __syncthreads();
  float acc = decb[j];
  for (int k = 0; k < 2 * HDIM; ++k) acc += cat[k] * decw[k * HDIM + j];
  gv[b * HDIM + j] = acc;
}

__global__ void dos_kernel(const float* __restrict__ emb, const float* __restrict__ gv,
                           const float* __restrict__ alphaP,
                           const float* __restrict__ w1, const float* __restrict__ b1,
                           const float* __restrict__ gam, const float* __restrict__ bet,
                           const float* __restrict__ aP,
                           const float* __restrict__ w2, const float* __restrict__ b2,
                           float* __restrict__ dos) {
  const int blk = blockIdx.x;
  const int e = blk / NGRAPH, b = blk - e * NGRAPH;
  const int j = threadIdx.x;
  __shared__ float z[HDIM];
  __shared__ float red[HDIM];
  z[j] = emb[e * HDIM + j] + alphaP[0] * gv[b * HDIM + j];
  __syncthreads();
  float acc = b1[j];
  for (int k = 0; k < HDIM; ++k) acc += z[k] * w1[k * HDIM + j];
  red[j] = acc;
  __syncthreads();
  for (int s = 64; s > 0; s >>= 1) { if (j < s) red[j] += red[j + s]; __syncthreads(); }
  const float mu = red[0] * (1.f / 128.f);
  __syncthreads();
  const float d = acc - mu;
  red[j] = d * d;
  __syncthreads();
  for (int s = 64; s > 0; s >>= 1) { if (j < s) red[j] += red[j + s]; __syncthreads(); }
  const float rs = rsqrtf(red[0] * (1.f / 128.f) + 1e-5f);
  __syncthreads();
  float h = (d * rs) * gam[j] + bet[j];
  const float pa = aP[0];
  h = h >= 0.f ? h : pa * h;
  red[j] = h * w2[j];
  __syncthreads();
  for (int s = 64; s > 0; s >>= 1) { if (j < s) red[j] += red[j + s]; __syncthreads(); }
  if (j == 0) dos[b * NEMB + e] = red[0] + b2[0];
}

__global__ void spark_kernel(const float* __restrict__ emb, const float* __restrict__ sw,
                             const float* __restrict__ sb, float* __restrict__ out) {
  const int i = blockIdx.x, j = threadIdx.x;
  __shared__ float red[HDIM];
  red[j] = emb[(i + 1) * HDIM + j] * sw[j] + emb[i * HDIM + j] * sw[HDIM + j];
  __syncthreads();
  for (int s = 64; s > 0; s >>= 1) { if (j < s) red[j] += red[j + s]; __syncthreads(); }
  const float v = 1.f / (1.f + __expf(-(red[0] + sb[0])));
  out[i * NGRAPH + j] = v;
}

// ---------------------------------------------------------------------------
extern "C" void kernel_launch(void* const* d_in, const int* in_sizes, int n_in,
                              void* d_out, int out_size, void* d_ws, size_t ws_size,
                              hipStream_t stream) {
  (void)in_sizes; (void)n_in; (void)out_size; (void)ws_size;
  int pi = 0;
  const float* emb  = (const float*)d_in[pi++];
  const float* enw1 = (const float*)d_in[pi++];
  const float* enb1 = (const float*)d_in[pi++];
  const float* ena  = (const float*)d_in[pi++];
  const float* enw2 = (const float*)d_in[pi++];
  const float* enb2 = (const float*)d_in[pi++];
  const float* eew1 = (const float*)d_in[pi++];
  const float* eeb1 = (const float*)d_in[pi++];
  const float* eea  = (const float*)d_in[pi++];
  const float* eew2 = (const float*)d_in[pi++];
  const float* eeb2 = (const float*)d_in[pi++];
  const float* egw1 = (const float*)d_in[pi++];
  const float* egb1 = (const float*)d_in[pi++];
  const float* ega  = (const float*)d_in[pi++];
  const float* egw2 = (const float*)d_in[pi++];
  const float* egb2 = (const float*)d_in[pi++];

  struct BlkP { const float *w1, *b1, *g, *bln, *a, *w2, *b2; };
  BlkP eB[3], nB[3];
  for (int l = 0; l < 3; ++l) {
    BlkP* arr[2] = { &eB[l], &nB[l] };
    for (int s = 0; s < 2; ++s) {
      BlkP* p = arr[s];
      p->w1  = (const float*)d_in[pi++];
      p->b1  = (const float*)d_in[pi++];
      p->g   = (const float*)d_in[pi++];
      p->bln = (const float*)d_in[pi++];
      p->a   = (const float*)d_in[pi++];
      p->w2  = (const float*)d_in[pi++];
      p->b2  = (const float*)d_in[pi++];
    }
  }
  const float* decw = (const float*)d_in[pi++];
  const float* decb = (const float*)d_in[pi++];
  BlkP oB;
  oB.w1  = (const float*)d_in[pi++];
  oB.b1  = (const float*)d_in[pi++];
  oB.g   = (const float*)d_in[pi++];
  oB.bln = (const float*)d_in[pi++];
  oB.a   = (const float*)d_in[pi++];
  oB.w2  = (const float*)d_in[pi++];
  oB.b2  = (const float*)d_in[pi++];
  const float* sparkw = (const float*)d_in[pi++];
  const float* sparkb = (const float*)d_in[pi++];
  const float* alphaG = (const float*)d_in[pi++];
  const float* xin    = (const float*)d_in[pi++];
  const float* eain   = (const float*)d_in[pi++];
  const float* globp  = (const float*)d_in[pi++];
  const int*   batch  = (const int*)d_in[pi++];
  const int*   eidx   = (const int*)d_in[pi++];

  // output regions: [dos | x | spark]
  float* dosOut   = (float*)d_out;
  float* xf       = dosOut + HDIM * NEMB;
  float* sparkOut = xf + (size_t)NNODES * HDIM;

  // workspace suballocation (256B aligned)
  char* wsp = (char*)d_ws;
  auto walloc = [&](size_t bytes) -> void* {
    void* p = (void*)wsp;
    wsp += (bytes + 255) & ~(size_t)255;
    return p;
  };
  bf16_t* xb     = (bf16_t*)walloc((size_t)NNODES * HDIM * 2);
  bf16_t* eab    = (bf16_t*)walloc((size_t)NEDGES * HDIM * 2);
  float*  agg    = (float*) walloc((size_t)NNODES * HDIM * 4);
  float*  uB     = (float*) walloc((size_t)NGRAPH * HDIM * 4);
  float*  pooled = (float*) walloc((size_t)NGRAPH * HDIM * 4);
  float*  gv     = (float*) walloc((size_t)NGRAPH * HDIM * 4);
  bf16_t* wEncN1 = (bf16_t*)walloc(96 * 128 * 2);
  bf16_t* wEncN2 = (bf16_t*)walloc(128 * 128 * 2);
  bf16_t* wEncE1 = (bf16_t*)walloc(64 * 128 * 2);
  bf16_t* wEncE2 = (bf16_t*)walloc(128 * 128 * 2);
  bf16_t *wE1[3], *wE2[3], *wN1[3], *wN2[3];
  for (int l = 0; l < 3; ++l) {
    wE1[l] = (bf16_t*)walloc(384 * 256 * 2);
    wE2[l] = (bf16_t*)walloc(256 * 128 * 2);
    wN1[l] = (bf16_t*)walloc(256 * 256 * 2);
    wN2[l] = (bf16_t*)walloc(256 * 128 * 2);
  }

  auto conv = [&](const float* src, bf16_t* dst, int K, int Nc, int Kpad) {
    const int total = Kpad * Nc;
    conv_weight_frag<<<(total + 255) / 256, 256, 0, stream>>>(src, dst, K, Nc, Kpad);
  };
  conv(enw1, wEncN1, 92, 128, 96);
  conv(enw2, wEncN2, 128, 128, 128);
  conv(eew1, wEncE1, 50, 128, 64);
  conv(eew2, wEncE2, 128, 128, 128);
  for (int l = 0; l < 3; ++l) {
    conv(eB[l].w1, wE1[l], 384, 256, 384);
    conv(eB[l].w2, wE2[l], 256, 128, 256);
    conv(nB[l].w1, wN1[l], 256, 256, 256);
    conv(nB[l].w2, wN2[l], 256, 128, 256);
  }

  // encoders
  enc_mlp_kernel<96, 92><<<NNODES / 16, 128, 0, stream>>>(
      xin, wEncN1, enb1, ena, wEncN2, enb2, xf, xb);
  enc_mlp_kernel<64, 50><<<NEDGES / 16, 128, 0, stream>>>(
      eain, wEncE1, eeb1, eea, wEncE2, eeb2, nullptr, eab);
  glob_enc_kernel<<<NGRAPH, HDIM, 0, stream>>>(globp, egw1, egb1, ega, egw2, egb2, uB);

  // message-passing layers
  const size_t aggN = (size_t)NNODES * HDIM;
  const int nodeBlocks = (NNODES + MT - 1) / MT;
  for (int l = 0; l < 3; ++l) {
    zero_f32<<<(int)((aggN + 255) / 256), 256, 0, stream>>>(agg, aggN);
    edge_block_kernel<<<NEDGES / MT, 256, 0, stream>>>(
        xb, eab, eidx, wE1[l], eB[l].b1, eB[l].g, eB[l].bln, eB[l].a,
        wE2[l], eB[l].b2, agg);
    node_block_kernel<<<nodeBlocks, 256, 0, stream>>>(
        xb, agg, wN1[l], nB[l].b1, nB[l].g, nB[l].bln, nB[l].a,
        wN2[l], nB[l].b2, xf);
  }

  // readout + heads
  zero_f32<<<(NGRAPH * HDIM + 255) / 256, 256, 0, stream>>>(pooled, NGRAPH * HDIM);
  pool_kernel<<<(int)((aggN + 255) / 256), 256, 0, stream>>>(xf, batch, pooled);
  graph_kernel<<<NGRAPH, HDIM, 0, stream>>>(uB, pooled, decw, decb, gv);
  dos_kernel<<<NEMB * NGRAPH, HDIM, 0, stream>>>(
      emb, gv, alphaG, oB.w1, oB.b1, oB.g, oB.bln, oB.a, oB.w2, oB.b2, dosOut);
  spark_kernel<<<NEMB - 1, HDIM, 0, stream>>>(emb, sparkw, sparkb, sparkOut);
}
